// GINNet_52063593562848
// MI455X (gfx1250) — matrix operand order repaired and checked by reading, not compiled, not run on hardware
//
#include <hip/hip_runtime.h>

// ---------------------------------------------------------------------------
// Types for CDNA5 WMMA (wave32): bf16 16x16x32, f32 accumulate
// ---------------------------------------------------------------------------
typedef __bf16 v16bf __attribute__((ext_vector_type(16)));
typedef float  v8f   __attribute__((ext_vector_type(8)));
typedef unsigned short us4 __attribute__((ext_vector_type(4)));

union Frag { us4 q[4]; v16bf v; };   // 16 bf16 = 8 VGPRs, assembled from 4x 8B pieces

__device__ __forceinline__ unsigned short bf16_bits(float f) {
    unsigned int u = __float_as_uint(f);
    unsigned int r = u + 0x7FFFu + ((u >> 16) & 1u);   // round-to-nearest-even
    return (unsigned short)(r >> 16);
}
__device__ __forceinline__ float bf16_to_f32(unsigned short s) {
    return __uint_as_float(((unsigned int)s) << 16);
}

struct NodePtrs { const float* p[9]; };
struct EdgePtrs { const float* p[5]; };

// ---------------------------------------------------------------------------
// Utility kernels
// ---------------------------------------------------------------------------
__global__ void k_zero_f32(float* __restrict__ p, int n) {
    int i = blockIdx.x * blockDim.x + threadIdx.x;
    if (i < n) p[i] = 0.0f;
}

// ---------------------------------------------------------------------------
// Pack head weight W[K,N] (f32) -> fragment-major bf16:
// tile (ct,kt) = 16 cols x 32 k; 512 elems contiguous; within tile:
// lane = {half,r}, elem i: W[kt*32 + (i&8)*2 + half*8 + (i&7)][ct*16 + r]  (ISA B layout)
// ---------------------------------------------------------------------------
__global__ void k_pack_b(const float* __restrict__ W, unsigned short* __restrict__ Wp,
                         int K, int N) {
    int t = blockIdx.x * blockDim.x + threadIdx.x;
    if (t >= K * N) return;
    int blk = t >> 9, idx = t & 511;
    int lane = idx >> 4, i = idx & 15;
    int ktiles = K >> 5;
    int ct = blk / ktiles, kt = blk - ct * ktiles;
    int half = lane >> 4, r = lane & 15;
    int kk  = kt * 32 + ((i & 8) << 1) + half * 8 + (i & 7);
    int col = ct * 16 + r;
    Wp[t] = bf16_bits(W[(size_t)kk * N + col]);
}

// ---------------------------------------------------------------------------
// Node embedding: 9 lookups (33) + x[:,9] + pos/10 -> h0 [N,37]
// ---------------------------------------------------------------------------
__global__ void k_node_embed(const float* __restrict__ x,
                             const float* __restrict__ pos,
                             NodePtrs tabs, float* __restrict__ h0, int N) {
    int n = blockIdx.x * blockDim.x + threadIdx.x;
    if (n >= N) return;
    const int nd[9] = {5, 2, 6, 3, 5, 4, 6, 1, 1};
    float* out = h0 + (size_t)n * 37;
    int o = 0;
#pragma unroll
    for (int t = 0; t < 9; ++t) {
        int idx = (int)x[(size_t)n * 13 + t];
        const float* row = tabs.p[t] + (size_t)idx * nd[t];
        for (int k = 0; k < nd[t]; ++k) out[o++] = row[k];
    }
    out[33] = x[(size_t)n * 13 + 9];
    out[34] = pos[(size_t)n * 3 + 0] * 0.1f;
    out[35] = pos[(size_t)n * 3 + 1] * 0.1f;
    out[36] = pos[(size_t)n * 3 + 2] * 0.1f;
}

// ---------------------------------------------------------------------------
// Scatter-sum: agg[dst] += h[src]   (segment_sum over edges)
// ---------------------------------------------------------------------------
__global__ void k_edge_agg(const float* __restrict__ h,
                           const int* __restrict__ src, const int* __restrict__ dst,
                           float* __restrict__ agg, int E, int dim) {
    int t = blockIdx.x * blockDim.x + threadIdx.x;
    if (t >= E * dim) return;
    int e = t / dim;
    int c = t - e * dim;
    atomicAdd(&agg[(size_t)dst[e] * dim + c], h[(size_t)src[e] * dim + c]);
}

// ---------------------------------------------------------------------------
// Y = relu((X [+Add]) @ W + b)  : small-dim conv linears (37/108), VALU path
// ---------------------------------------------------------------------------
__global__ void k_lin_relu(const float* __restrict__ X, const float* __restrict__ Add,
                           const float* __restrict__ W, const float* __restrict__ Bv,
                           float* __restrict__ Y, int M, int din, int dout) {
    int t = blockIdx.x * blockDim.x + threadIdx.x;
    if (t >= M * dout) return;
    int m = t / dout;
    int c = t - m * dout;
    float acc = Bv[c];
    const float* xr = X + (size_t)m * din;
    const float* ar = Add ? Add + (size_t)m * din : nullptr;
    for (int i = 0; i < din; ++i) {
        float xv = xr[i];
        if (ar) xv += ar[i];
        acc = fmaf(xv, W[(size_t)i * dout + c], acc);
    }
    Y[t] = fmaxf(acc, 0.0f);
}

// ---------------------------------------------------------------------------
// BatchNorm: batch stats (biased var), eps=1e-5
// ---------------------------------------------------------------------------
__global__ void k_bn_stats(const float* __restrict__ H, float* __restrict__ stats,
                           int M, int dim) {
    int c = blockIdx.x;
    float s = 0.f, s2 = 0.f;
    for (int m = blockIdx.y * blockDim.x + threadIdx.x; m < M;
         m += gridDim.y * blockDim.x) {
        float v = H[(size_t)m * dim + c];
        s += v; s2 += v * v;
    }
    __shared__ float sh[256], sh2[256];
    sh[threadIdx.x] = s; sh2[threadIdx.x] = s2;
    __syncthreads();
    for (int w = 128; w > 0; w >>= 1) {
        if ((int)threadIdx.x < w) {
            sh[threadIdx.x]  += sh[threadIdx.x + w];
            sh2[threadIdx.x] += sh2[threadIdx.x + w];
        }
        __syncthreads();
    }
    if (threadIdx.x == 0) {
        atomicAdd(&stats[c], sh[0]);
        atomicAdd(&stats[dim + c], sh2[0]);
    }
}

__global__ void k_bn_apply(float* __restrict__ H, const float* __restrict__ stats,
                           const float* __restrict__ g, const float* __restrict__ b,
                           int M, int dim) {
    int t = blockIdx.x * blockDim.x + threadIdx.x;
    if (t >= M * dim) return;
    int c = t % dim;
    float inv = 1.0f / (float)M;
    float mu = stats[c] * inv;
    float var = stats[dim + c] * inv - mu * mu;
    H[t] = g[c] * (H[t] - mu) * rsqrtf(var + 1e-5f) + b[c];
}

// ---------------------------------------------------------------------------
// Per-edge: embed + MLP 11->22->40->40, concat [h2[src], h2[dst], e40] -> bf16
// ---------------------------------------------------------------------------
__global__ void k_edge_prep(const float* __restrict__ eattr, EdgePtrs tabs,
                            const float* __restrict__ w0, const float* __restrict__ b0,
                            const float* __restrict__ w1, const float* __restrict__ b1,
                            const float* __restrict__ w2, const float* __restrict__ b2,
                            const int* __restrict__ src, const int* __restrict__ dst,
                            const float* __restrict__ h2,
                            unsigned short* __restrict__ xe, int e0, int cnt) {
    int t = blockIdx.x * blockDim.x + threadIdx.x;
    if (t >= cnt) return;
    int e = e0 + t;
    const int ed[5] = {3, 2, 3, 1, 1};
    float e11[11];
    int o = 0;
#pragma unroll
    for (int i = 0; i < 5; ++i) {
        int idx = (int)eattr[(size_t)e * 6 + i];
        const float* row = tabs.p[i] + (size_t)idx * ed[i];
        for (int k = 0; k < ed[i]; ++k) e11[o++] = row[k];
    }
    e11[10] = eattr[(size_t)e * 6 + 5] * 0.1f;

    float a1[22];
#pragma unroll
    for (int j = 0; j < 22; ++j) {
        float acc = b0[j];
        for (int i = 0; i < 11; ++i) acc = fmaf(e11[i], w0[i * 22 + j], acc);
        a1[j] = fmaxf(acc, 0.f);
    }
    float a2[40];
#pragma unroll
    for (int j = 0; j < 40; ++j) {
        float acc = b1[j];
        for (int i = 0; i < 22; ++i) acc = fmaf(a1[i], w1[i * 40 + j], acc);
        a2[j] = fmaxf(acc, 0.f);
    }
    float a3[40];
#pragma unroll
    for (int j = 0; j < 40; ++j) {
        float acc = b2[j];
        for (int i = 0; i < 40; ++i) acc = fmaf(a2[i], w2[i * 40 + j], acc);
        a3[j] = acc;                       // no relu after last edge linear
    }

    unsigned short* row = xe + (size_t)t * 256;
    int s = src[e], d = dst[e];
    for (int c = 0; c < 108; ++c) row[c]       = bf16_bits(h2[(size_t)s * 108 + c]);
    for (int c = 0; c < 108; ++c) row[108 + c] = bf16_bits(h2[(size_t)d * 108 + c]);
    for (int c = 0; c < 40;  ++c) row[216 + c] = bf16_bits(a3[c]);
}

// ---------------------------------------------------------------------------
// WMMA GEMM:  Y[M,N] = relu(A[M,K] @ W[K,N] + bias)   (bf16 in, f32 acc, bf16 out)
// Block = 256 threads = 8 waves (2x4). Block tile 64x128; each wave owns a
// 32x32 region = 2x2 WMMA tiles -> 4 v_wmma per K-step.
// A: LDS-staged (coalesced, reused by 4 waves), branchless row guard.
// B: NO LDS - loaded per-wave from fragment-major packed weights (Wp) with
//    two global_load_b128 per fragment, lane-contiguous; hot in L2.
// ---------------------------------------------------------------------------
__global__ void k_wmma_gemm(const unsigned short* __restrict__ A,
                            const unsigned short* __restrict__ Wp,
                            const float* __restrict__ bias,
                            unsigned short* __restrict__ Y,
                            int M, int N, int K) {
    __shared__ unsigned short As[64][36];    // [row][k]  row stride 72B (8B aligned)

    const int tid  = threadIdx.x;
    const int wave = tid >> 5;
    const int lane = tid & 31;
    const int half = lane >> 4;
    const int r    = lane & 15;
    const int wm   = wave >> 2;              // 0..1  wave row
    const int wn   = wave & 3;               // 0..3  wave col
    const int rowBase = blockIdx.x * 64;
    const int colBase = blockIdx.y * 128;
    const int ktiles  = K >> 5;

    // A staging: per-thread constants (row guard is loop-invariant)
    const int ai  = tid * 8;                 // 256*8 == 64*32
    const int arr = ai >> 5, acol = ai & 31;
    const int agr = rowBase + arr;
    const bool avalid = (agr < M);
    const unsigned short* aRow = A + (size_t)(avalid ? agr : 0) * K;  // clamped, safe
    const us4 zz = {0, 0, 0, 0};

    // B: per-wave fragment-major pointers (two 16-col tiles per wave)
    const int ct0 = (colBase >> 4) + wn * 2;
    const unsigned short* bB0 = Wp + ((size_t)ct0 * ktiles) * 512 + lane * 16;
    const unsigned short* bB1 = Wp + ((size_t)(ct0 + 1) * ktiles) * 512 + lane * 16;

    v8f acc00 = {}, acc01 = {}, acc10 = {}, acc11 = {};

    for (int kt = 0; kt < ktiles; ++kt) {
        const int k0 = kt * 32;
        // --- stage A tile 64x32 (branchless: clamped row + select-zero) ---
        {
            us4 v0 = *(const us4*)(aRow + k0 + acol);
            us4 v1 = *(const us4*)(aRow + k0 + acol + 4);
            v0 = avalid ? v0 : zz;
            v1 = avalid ? v1 : zz;
            *(us4*)&As[arr][acol]     = v0;
            *(us4*)&As[arr][acol + 4] = v1;
        }
        // --- B fragments straight from packed global (no LDS) ---
        Frag b0, b1;
        {
            const us4* p0 = (const us4*)(bB0 + (size_t)kt * 512);
            const us4* p1 = (const us4*)(bB1 + (size_t)kt * 512);
            b0.q[0] = p0[0]; b0.q[1] = p0[1]; b0.q[2] = p0[2]; b0.q[3] = p0[3];
            b1.q[0] = p1[0]; b1.q[1] = p1[1]; b1.q[2] = p1[2]; b1.q[3] = p1[3];
        }
        // prefetch next K-tile of packed weights (global_prefetch_b8)
        if (kt + 1 < ktiles) {
            __builtin_prefetch(bB0 + (size_t)(kt + 1) * 512, 0, 1);
            __builtin_prefetch(bB1 + (size_t)(kt + 1) * 512, 0, 1);
        }
        __syncthreads();

        // --- A fragments from LDS (ISA 16-bit A 16x32 layout):
        //     lane {half,r}: 8 bf16 at K=half*8, 8 bf16 at K=16+half*8 ---
        Frag a0, a1;
        {
            const unsigned short* r0 = &As[wm * 32 + r][0];
            const unsigned short* r1 = &As[wm * 32 + 16 + r][0];
            a0.q[0] = *(const us4*)(r0 + half * 8);
            a0.q[1] = *(const us4*)(r0 + half * 8 + 4);
            a0.q[2] = *(const us4*)(r0 + 16 + half * 8);
            a0.q[3] = *(const us4*)(r0 + 16 + half * 8 + 4);
            a1.q[0] = *(const us4*)(r1 + half * 8);
            a1.q[1] = *(const us4*)(r1 + half * 8 + 4);
            a1.q[2] = *(const us4*)(r1 + 16 + half * 8);
            a1.q[3] = *(const us4*)(r1 + 16 + half * 8 + 4);
        }
        acc00 = __builtin_amdgcn_wmma_f32_16x16x32_bf16(false, a0.v, false, b0.v, (short)0, acc00, false, false);
        acc01 = __builtin_amdgcn_wmma_f32_16x16x32_bf16(false, a0.v, false, b1.v, (short)0, acc01, false, false);
        acc10 = __builtin_amdgcn_wmma_f32_16x16x32_bf16(false, a1.v, false, b0.v, (short)0, acc10, false, false);
        acc11 = __builtin_amdgcn_wmma_f32_16x16x32_bf16(false, a1.v, false, b1.v, (short)0, acc11, false, false);
        __syncthreads();
    }

    // --- store 2x2 tiles: C/D layout M = half*8 + d, N = lane&15 ---
#pragma unroll
    for (int sm = 0; sm < 2; ++sm) {
#pragma unroll
        for (int sn = 0; sn < 2; ++sn) {
            v8f acc = (sm == 0) ? ((sn == 0) ? acc00 : acc01)
                                : ((sn == 0) ? acc10 : acc11);
#pragma unroll
            for (int d = 0; d < 8; ++d) {
                int grow = rowBase + wm * 32 + sm * 16 + half * 8 + d;
                int gcol = colBase + wn * 32 + sn * 16 + r;
                if (grow < M) {
                    float v = fmaxf(acc[d] + bias[gcol], 0.0f);
                    Y[(size_t)grow * N + gcol] = bf16_bits(v);
                }
            }
        }
    }
}

// ---------------------------------------------------------------------------
// Final: per-edge dot(y3[256], w3) + b3, atomically accumulated per graph
// ---------------------------------------------------------------------------
__global__ void k_head_final(const unsigned short* __restrict__ y3,
                             const float* __restrict__ w3, const float* __restrict__ b3,
                             const int* __restrict__ ebatch, float* __restrict__ out,
                             int e0, int cnt) {
    int t = blockIdx.x * blockDim.x + threadIdx.x;
    if (t >= cnt) return;
    float acc = b3[0];
    const unsigned short* row = y3 + (size_t)t * 256;
    for (int i = 0; i < 256; ++i) acc = fmaf(bf16_to_f32(row[i]), w3[i], acc);
    atomicAdd(&out[ebatch[e0 + t]], acc);
}

// ---------------------------------------------------------------------------
// Orchestration
// ---------------------------------------------------------------------------
extern "C" void kernel_launch(void* const* d_in, const int* in_sizes, int n_in,
                              void* d_out, int out_size, void* d_ws, size_t ws_size,
                              hipStream_t stream) {
    const int N = 100000, E = 400000, Bg = 512, DIM = 108;
    const int CHUNK = 32768;

    const float* x      = (const float*)d_in[0];
    const float* pos    = (const float*)d_in[1];
    const int*   eidx   = (const int*)d_in[2];
    const float* eattr  = (const float*)d_in[3];
    const int*   ebatch = (const int*)d_in[4];
    NodePtrs ntab; for (int i = 0; i < 9; ++i) ntab.p[i] = (const float*)d_in[5 + i];
    EdgePtrs etab; for (int i = 0; i < 5; ++i) etab.p[i] = (const float*)d_in[14 + i];
    const float* c1w0 = (const float*)d_in[19]; const float* c1b0 = (const float*)d_in[20];
    const float* c1w1 = (const float*)d_in[21]; const float* c1b1 = (const float*)d_in[22];
    const float* c2w0 = (const float*)d_in[23]; const float* c2b0 = (const float*)d_in[24];
    const float* c2w1 = (const float*)d_in[25]; const float* c2b1 = (const float*)d_in[26];
    const float* bn1g = (const float*)d_in[27]; const float* bn1b = (const float*)d_in[28];
    const float* bn2g = (const float*)d_in[29]; const float* bn2b = (const float*)d_in[30];
    const float* ew0 = (const float*)d_in[31]; const float* eb0 = (const float*)d_in[32];
    const float* ew1 = (const float*)d_in[33]; const float* eb1 = (const float*)d_in[34];
    const float* ew2 = (const float*)d_in[35]; const float* eb2 = (const float*)d_in[36];
    const float* hw0 = (const float*)d_in[37]; const float* hb0 = (const float*)d_in[38];
    const float* hw1 = (const float*)d_in[39]; const float* hb1 = (const float*)d_in[40];
    const float* hw2 = (const float*)d_in[41]; const float* hb2 = (const float*)d_in[42];
    const float* hw3 = (const float*)d_in[43]; const float* hb3 = (const float*)d_in[44];

    const int* src = eidx;
    const int* dst = eidx + E;
    float* out = (float*)d_out;

    // Workspace carve (256B aligned)
    char* base = (char*)d_ws;
    size_t off = 0;
    auto carve = [&](size_t bytes) -> char* {
        char* p = base + off;
        off = (off + bytes + 255) & ~(size_t)255;
        return p;
    };
    float* h0   = (float*)carve((size_t)N * 37 * 4);
    float* agg  = (float*)carve((size_t)N * DIM * 4);
    float* tmp  = (float*)carve((size_t)N * DIM * 4);
    float* h1   = (float*)carve((size_t)N * DIM * 4);
    float* h2   = (float*)carve((size_t)N * DIM * 4);
    float* stats = (float*)carve((size_t)2 * DIM * 4);
    unsigned short* w0b = (unsigned short*)carve((size_t)256 * 512 * 2);
    unsigned short* w1b = (unsigned short*)carve((size_t)512 * 512 * 2);
    unsigned short* w2b = (unsigned short*)carve((size_t)512 * 256 * 2);
    unsigned short* xe = (unsigned short*)carve((size_t)CHUNK * 256 * 2);
    unsigned short* y1 = (unsigned short*)carve((size_t)CHUNK * 512 * 2);
    unsigned short* y2 = (unsigned short*)carve((size_t)CHUNK * 512 * 2);
    unsigned short* y3 = (unsigned short*)carve((size_t)CHUNK * 256 * 2);

    auto gz = [](int n) { return (n + 255) / 256; };

    // Output + aggregation buffers zeroed (harness does not re-zero between replays)
    k_zero_f32<<<gz(Bg), 256, 0, stream>>>(out, Bg);
    k_zero_f32<<<gz(N * 37), 256, 0, stream>>>(agg, N * 37);

    // Node embedding -> h0 [N,37]
    k_node_embed<<<gz(N), 256, 0, stream>>>(x, pos, ntab, h0, N);

    // ---- GIN conv1 ----
    k_edge_agg<<<gz(E * 37), 256, 0, stream>>>(h0, src, dst, agg, E, 37);
    k_lin_relu<<<gz(N * DIM), 256, 0, stream>>>(h0, agg, c1w0, c1b0, tmp, N, 37, DIM);
    k_lin_relu<<<gz(N * DIM), 256, 0, stream>>>(tmp, nullptr, c1w1, c1b1, h1, N, DIM, DIM);
    k_zero_f32<<<gz(2 * DIM), 256, 0, stream>>>(stats, 2 * DIM);
    { dim3 g(DIM, 64); k_bn_stats<<<g, 256, 0, stream>>>(h1, stats, N, DIM); }
    k_bn_apply<<<gz(N * DIM), 256, 0, stream>>>(h1, stats, bn1g, bn1b, N, DIM);

    // ---- GIN conv2 ----
    k_zero_f32<<<gz(N * DIM), 256, 0, stream>>>(agg, N * DIM);
    k_edge_agg<<<gz(E * DIM), 256, 0, stream>>>(h1, src, dst, agg, E, DIM);
    k_lin_relu<<<gz(N * DIM), 256, 0, stream>>>(h1, agg, c2w0, c2b0, tmp, N, DIM, DIM);
    k_lin_relu<<<gz(N * DIM), 256, 0, stream>>>(tmp, nullptr, c2w1, c2b1, h2, N, DIM, DIM);
    k_zero_f32<<<gz(2 * DIM), 256, 0, stream>>>(stats, 2 * DIM);
    { dim3 g(DIM, 64); k_bn_stats<<<g, 256, 0, stream>>>(h2, stats, N, DIM); }
    k_bn_apply<<<gz(N * DIM), 256, 0, stream>>>(h2, stats, bn2g, bn2b, N, DIM);

    // ---- Head weights -> fragment-major bf16 (once per launch; deterministic) ----
    k_pack_b<<<gz(256 * 512), 256, 0, stream>>>(hw0, w0b, 256, 512);
    k_pack_b<<<gz(512 * 512), 256, 0, stream>>>(hw1, w1b, 512, 512);
    k_pack_b<<<gz(512 * 256), 256, 0, stream>>>(hw2, w2b, 512, 256);

    // ---- Head: chunked over edges, WMMA GEMMs ----
    for (int e0 = 0; e0 < E; e0 += CHUNK) {
        int cnt = (E - e0 < CHUNK) ? (E - e0) : CHUNK;
        k_edge_prep<<<gz(cnt), 256, 0, stream>>>(eattr, etab, ew0, eb0, ew1, eb1,
                                                 ew2, eb2, src, dst, h2, xe, e0, cnt);
        int mt = (cnt + 63) / 64;
        { dim3 g(mt, 512 / 128); k_wmma_gemm<<<g, 256, 0, stream>>>(xe, w0b, hb0, y1, cnt, 512, 256); }
        { dim3 g(mt, 512 / 128); k_wmma_gemm<<<g, 256, 0, stream>>>(y1, w1b, hb1, y2, cnt, 512, 512); }
        { dim3 g(mt, 256 / 128); k_wmma_gemm<<<g, 256, 0, stream>>>(y2, w2b, hb2, y3, cnt, 256, 512); }
        k_head_final<<<gz(cnt), 256, 0, stream>>>(y3, hw3, hb3, ebatch, out, e0, cnt);
    }
}
